// CSSM_32590211842136
// MI455X (gfx1250) — compile-verified, matrix operand order checked by reading, not run on hardware
//
#include <hip/hip_runtime.h>
#include <stdint.h>

typedef __attribute__((ext_vector_type(16))) __bf16 v16bf;
typedef __attribute__((ext_vector_type(8)))  float  v8f;
typedef unsigned short u16;
typedef unsigned int   u32;

// ---------- helpers ----------
__device__ __forceinline__ u16 f2bf(float f) {
  u32 u = __float_as_uint(f);
  u32 r = u + 0x7FFFu + ((u >> 16) & 1u);     // round-to-nearest-even
  return (u16)(r >> 16);
}
__device__ __forceinline__ __bf16 us_as_bf(u16 s) {
  __bf16 h; __builtin_memcpy(&h, &s, 2); return h;
}
__device__ __forceinline__ float dsilu(float v) { return v / (1.f + __expf(-v)); }

// ---------- elementwise ----------
__global__ void k_f32_to_bf16(const float* __restrict__ in, u16* __restrict__ out, long n) {
  long i = (long)blockIdx.x * blockDim.x + threadIdx.x;
  if (i < n) out[i] = f2bf(in[i]);
}
__global__ void k_add(const float* __restrict__ a, const float* __restrict__ b,
                      float* __restrict__ o, long n) {
  long i = (long)blockIdx.x * blockDim.x + threadIdx.x;
  if (i < n) o[i] = a[i] + b[i];
}
// out[b,c,n] = g[b,n,c] + x12[b,n,c]   (final fuse + residual, transposed store)
__global__ void k_final(const float* __restrict__ g, const float* __restrict__ x12,
                        float* __restrict__ out, int Bb, int Cch, int N) {
  long i = (long)blockIdx.x * blockDim.x + threadIdx.x;
  long tot = (long)Bb * N * Cch;
  if (i >= tot) return;
  int c = (int)(i % Cch);
  long bn = i / Cch;
  long n = bn % N, b = bn / N;
  out[(b * Cch + c) * (long)N + n] = g[i] + x12[i];
}

// ---------- WMMA GEMM:  O[m,n] = sum_k A[m,k]*W[n,k]  (both row-major [row][k], bf16) ----------
__global__ void k_wmma_gemm(const u16* __restrict__ A, int lda, long sA,
                            const u16* __restrict__ Bw, int ldb, long sB,
                            float* __restrict__ O, int ldc, long sC,
                            int tilesN, int K, const float* __restrict__ bias) {
  int wave = blockIdx.x;
  int tm = (wave / tilesN) << 4;
  int tn = (wave % tilesN) << 4;
  int b = blockIdx.y;
  int lane = threadIdx.x;
  int row = lane & 15, half = lane >> 4;
  const u16* Ab = A + (long)b * sA + (long)(tm + row) * lda;
  const u16* Bb = Bw + (long)b * sB + (long)(tn + row) * ldb;
  v8f acc = {};
  for (int k0 = 0; k0 < K; k0 += 32) {
    if (k0 + 32 < K) {
      __builtin_prefetch(Ab + k0 + 32, 0, 1);   // global_prefetch_b8
      __builtin_prefetch(Bb + k0 + 32, 0, 1);
    }
    v16bf fa, fb;
#pragma unroll
    for (int r = 0; r < 8; ++r) {
      // 16-bit A 16x32 lane layout: lanes 0-15 K=0..7,16..23 ; lanes 16-31 K=8..15,24..31
      int k = k0 + 2 * (r & 3) + ((r & 4) ? 16 : 0) + 8 * half;
      u32 wa = *(const u32*)(Ab + k);
      u32 wb = *(const u32*)(Bb + k);
      fa[2 * r]     = us_as_bf((u16)(wa & 0xffffu));
      fa[2 * r + 1] = us_as_bf((u16)(wa >> 16));
      fb[2 * r]     = us_as_bf((u16)(wb & 0xffffu));
      fb[2 * r + 1] = us_as_bf((u16)(wb >> 16));
    }
    acc = __builtin_amdgcn_wmma_f32_16x16x32_bf16(false, fa, false, fb,
                                                  (short)0, acc, false, false);
  }
  float* Ob = O + (long)b * sC;
  int n = tn + row;
#pragma unroll
  for (int r = 0; r < 8; ++r) {          // D layout: VGPR r -> M = r (+8 for lanes 16-31)
    int m = tm + r + 8 * half;
    float v = acc[r];
    if (bias) v += bias[n];
    Ob[(long)m * ldc + n] = v;
  }
}

// ---------- small GEMM (tiny N or K; f32), act: 0 none, 1 softplus ----------
__global__ void k_gemm_small(const float* __restrict__ A, int lda,
                             const float* __restrict__ W, int ldw,
                             const float* __restrict__ bias,
                             float* __restrict__ O, int ldo,
                             long M, int N, int K, int act) {
  long idx = (long)blockIdx.x * blockDim.x + threadIdx.x;
  if (idx >= M * (long)N) return;
  long m = idx / N; int n = (int)(idx % N);
  const float* a = A + m * lda;
  const float* w = W + (long)n * ldw;
  float s = bias ? bias[n] : 0.f;
  for (int k = 0; k < K; ++k) s += a[k] * w[k];
  if (act == 1) s = (s > 20.f) ? s : __logf(1.f + __expf(s));
  O[m * ldo + n] = s;
}

// ---------- LDS tiled transpose: out[b][i][r] = in[b][r][i] ----------
__global__ void k_transpose(const float* __restrict__ in, int R, int Cc,
                            float* __restrict__ out_f, int ldo_f,
                            u16* __restrict__ out_b, int ldo_b, int co_b) {
  __shared__ float tile[16][17];
  int b = blockIdx.z;
  int r0 = blockIdx.x * 16, c0 = blockIdx.y * 16;
  int tx = threadIdx.x, ty = threadIdx.y;
  int r = r0 + ty, c = c0 + tx;
  float v = 0.f;
  if (r < R && c < Cc) v = in[((long)b * R + r) * Cc + c];
  tile[ty][tx] = v;
  __syncthreads();
  int ro = c0 + ty, cw = r0 + tx;
  if (ro < Cc && cw < R) {
    float t = tile[tx][ty];
    if (out_f) out_f[((long)b * Cc + ro) * ldo_f + cw] = t;
    if (out_b) out_b[((long)b * Cc + ro) * ldo_b + co_b + cw] = f2bf(t);
  }
}

// ---------- LayerNorm over last dim (W<=128), optional residual, f32/bf16 outs ----------
__global__ void k_layernorm(const float* __restrict__ in, const float* __restrict__ res,
                            const float* __restrict__ g, const float* __restrict__ bta,
                            float* __restrict__ out_f, u16* __restrict__ out_b, int W) {
  __shared__ float red[128];
  long row = blockIdx.x;
  int t = threadIdx.x;
  float x = 0.f;
  if (t < W) {
    x = in[row * W + t];
    if (res) x += res[row * W + t];
  }
  red[t] = x; __syncthreads();
  for (int s = 64; s > 0; s >>= 1) { if (t < s) red[t] += red[t + s]; __syncthreads(); }
  float mean = red[0] / W; __syncthreads();
  float d = (t < W) ? (x - mean) : 0.f;
  red[t] = d * d; __syncthreads();
  for (int s = 64; s > 0; s >>= 1) { if (t < s) red[t] += red[t + s]; __syncthreads(); }
  float inv = rsqrtf(red[0] / W + 1e-5f);
  if (t < W) {
    float o = d * inv * g[t] + bta[t];
    if (out_f) out_f[row * W + t] = o;
    if (out_b) out_b[row * W + t] = f2bf(o);
  }
}

// ---------- causal depthwise conv (taps=4) + SiLU; channel-last ----------
__global__ void k_dwconv_silu(const float* __restrict__ xin, int ldin,
                              const float* __restrict__ cw, const float* __restrict__ cb,
                              float* __restrict__ xc, u16* __restrict__ xcb,
                              int Bb, int L, int di) {
  int t = blockIdx.x * blockDim.x + threadIdx.x;
  if (t >= Bb * di) return;
  int b = t / di, d = t % di;
  float w0 = cw[d * 4 + 0], w1 = cw[d * 4 + 1], w2 = cw[d * 4 + 2], w3 = cw[d * 4 + 3];
  float bias = cb[d];
  float x0 = 0.f, x1 = 0.f, x2 = 0.f;
  const float* inb = xin + (long)b * L * ldin + d;
  float* ob = xc + (long)b * L * di + d;
  u16* obb = xcb + (long)b * L * di + d;
  for (int l = 0; l < L; ++l) {
    float x3 = inb[(long)l * ldin];
    float a = bias + w0 * x0 + w1 * x1 + w2 * x2 + w3 * x3;
    a = dsilu(a);
    ob[(long)l * di] = a;
    obb[(long)l * di] = f2bf(a);
    x0 = x1; x1 = x2; x2 = x3;
  }
}

// ---------- fused selective scan:  y = (scan(h) + x*D) * silu(z) ----------
__global__ void k_scan(const float* __restrict__ dt, const float* __restrict__ xdbl,
                       int ldx, int offB,
                       const float* __restrict__ xc, const float* __restrict__ z, int ldz,
                       const float* __restrict__ Alog, const float* __restrict__ Dp,
                       float* __restrict__ y, int Bb, int L, int di) {
  int t = blockIdx.x * blockDim.x + threadIdx.x;
  if (t >= Bb * di) return;
  int b = t / di, d = t % di;
  float A[8], h[8];
#pragma unroll
  for (int s = 0; s < 8; ++s) { A[s] = -__expf(Alog[d * 8 + s]); h[s] = 0.f; }
  float Dd = Dp[d];
  long rb = (long)b * L;
  for (int l = 0; l < L; ++l) {
    long r = rb + l;
    float dtv = dt[r * di + d];
    float xv  = xc[r * di + d];
    float zv  = z[r * ldz + d];
    const float* xd = xdbl + r * ldx + offB;   // [B(8) | C(8)]
    float acc = 0.f;
#pragma unroll
    for (int s = 0; s < 8; ++s) {
      h[s] = __expf(dtv * A[s]) * h[s] + dtv * xd[s] * xv;
      acc += h[s] * xd[8 + s];
    }
    y[r * di + d] = (acc + xv * Dd) * dsilu(zv);
  }
}

// ---------- instance-norm stats per (b,c) over len ----------
__global__ void k_inorm_stats(const float* __restrict__ x, int len,
                              float* __restrict__ mean, float* __restrict__ var) {
  __shared__ float s1[256], s2[256];
  int bc = blockIdx.x; int t = threadIdx.x;
  const float* p = x + (long)bc * len;
  float a = 0.f, b = 0.f;
  for (int i = t; i < len; i += 256) { float v = p[i]; a += v; b += v * v; }
  s1[t] = a; s2[t] = b; __syncthreads();
  for (int s = 128; s > 0; s >>= 1) {
    if (t < s) { s1[t] += s1[t + s]; s2[t] += s2[t + s]; } __syncthreads();
  }
  if (t == 0) { float m = s1[0] / len; mean[bc] = m; var[bc] = s2[0] / len - m * m; }
}
// batch var of instance-normed data: mean_b( v/(v+1e-3) ) ; batch mean == 0 analytically
__global__ void k_bnorm_var(const float* __restrict__ var, int Bb, int Cch,
                            float* __restrict__ bv) {
  int c = blockIdx.x * blockDim.x + threadIdx.x;
  if (c >= Cch) return;
  float s = 0.f;
  for (int b = 0; b < Bb; ++b) { float v = var[b * Cch + c]; s += v / (v + 1e-3f); }
  bv[c] = s / Bb;
}
// apply IN -> BN -> ReLU, write position-major bf16 (B*len, C)
__global__ void k_in_bn_relu(const float* __restrict__ x, int len, int Cch,
                             const float* __restrict__ mean, const float* __restrict__ var,
                             const float* __restrict__ bv,
                             const float* __restrict__ gw, const float* __restrict__ gb,
                             u16* __restrict__ out) {
  long pos = blockIdx.x;                 // b*len + n
  int c = threadIdx.x;                   // blockDim == Cch
  long b = pos / len, n = pos % len;
  float m = mean[b * Cch + c], v = var[b * Cch + c];
  float t = (x[(b * Cch + c) * (long)len + n] - m) * rsqrtf(v + 1e-3f);
  t = t * rsqrtf(bv[c] + 1e-5f) * gw[c] + gb[c];
  out[pos * Cch + c] = f2bf(t > 0.f ? t : 0.f);
}

// ---------- strided softmax: element i of row r at base(r) + i*estride ----------
__global__ void k_softmax(float* __restrict__ io, long rstrideA, int rmod, long rstrideB,
                          int len, int estride) {
  __shared__ float red[256];
  long r = blockIdx.x; int t = threadIdx.x;
  float* p = io + (r / rmod) * rstrideA + (r % rmod) * rstrideB;
  float mx = -1e30f;
  for (int i = t; i < len; i += 256) mx = fmaxf(mx, p[(long)i * estride]);
  red[t] = mx; __syncthreads();
  for (int s = 128; s > 0; s >>= 1) { if (t < s) red[t] = fmaxf(red[t], red[t + s]); __syncthreads(); }
  mx = red[0]; __syncthreads();
  float sm = 0.f;
  for (int i = t; i < len; i += 256) {
    float e = __expf(p[(long)i * estride] - mx);
    p[(long)i * estride] = e; sm += e;
  }
  red[t] = sm; __syncthreads();
  for (int s = 128; s > 0; s >>= 1) { if (t < s) red[t] += red[t + s]; __syncthreads(); }
  float inv = 1.f / red[0];
  for (int i = t; i < len; i += 256) p[(long)i * estride] *= inv;
}

// =====================================================================================
extern "C" void kernel_launch(void* const* d_in, const int* in_sizes, int n_in,
                              void* d_out, int out_size, void* d_ws, size_t ws_size,
                              hipStream_t stream) {
  (void)in_sizes; (void)n_in; (void)out_size; (void)ws_size;
  constexpr int Bb = 8, C = 128, N = 4096, K = 64;
  constexpr int DI = 256, DR = 8;                 // big Mamba (d_model=C)
  constexpr int DM2 = 64, DI2 = 128, DR2 = 4, L2 = C; // SAM (d_model=K)
  const long M  = (long)Bb * N;                   // 32768
  const long M2 = (long)Bb * C;                   // 1024

  // ---- inputs (setup_inputs dict insertion order, recursively flattened) ----
  const float* data       = (const float*)d_in[0];
  const float* mb_ln_w    = (const float*)d_in[1];
  const float* mb_ln_b    = (const float*)d_in[2];
  const float* mm_in_w    = (const float*)d_in[3];   // (512,128)
  const float* mm_conv_w  = (const float*)d_in[4];   // (256,1,4)
  const float* mm_conv_b  = (const float*)d_in[5];
  const float* mm_xproj_w = (const float*)d_in[6];   // (24,256)
  const float* mm_dtproj_w= (const float*)d_in[7];   // (256,8)
  const float* mm_dtproj_b= (const float*)d_in[8];
  const float* mm_Alog    = (const float*)d_in[9];   // (256,8)
  const float* mm_D       = (const float*)d_in[10];
  const float* mm_out_w   = (const float*)d_in[11];  // (128,256)
  const float* dn_bnw = (const float*)d_in[12];
  const float* dn_bnb = (const float*)d_in[13];
  const float* dn_w   = (const float*)d_in[14];      // (64,128)
  const float* dn_b   = (const float*)d_in[15];
  const float* up_bnw = (const float*)d_in[16];
  const float* up_bnb = (const float*)d_in[17];
  const float* up_w   = (const float*)d_in[18];      // (64,128)
  const float* up_b   = (const float*)d_in[19];
  const float* c1_bnw = (const float*)d_in[20];
  const float* c1_bnb = (const float*)d_in[21];
  const float* c1_w   = (const float*)d_in[22];      // (128,128)
  const float* c1_b   = (const float*)d_in[23];
  const float* cs_lnw = (const float*)d_in[24];      // (64)
  const float* cs_lnb = (const float*)d_in[25];
  const float* sm_in_w    = (const float*)d_in[26];  // (256,64)
  const float* sm_conv_w  = (const float*)d_in[27];  // (128,1,4)
  const float* sm_conv_b  = (const float*)d_in[28];
  const float* sm_xproj_w = (const float*)d_in[29];  // (20,128)
  const float* sm_dtproj_w= (const float*)d_in[30];  // (128,4)
  const float* sm_dtproj_b= (const float*)d_in[31];
  const float* sm_Alog    = (const float*)d_in[32];  // (128,8)
  const float* sm_D       = (const float*)d_in[33];
  const float* sm_out_w   = (const float*)d_in[34];  // (64,128)
  const float* c3_bnw = (const float*)d_in[35];
  const float* c3_bnb = (const float*)d_in[36];
  const float* c3_w   = (const float*)d_in[37];      // (128,128)
  const float* c3_b   = (const float*)d_in[38];
  const float* shot_w = (const float*)d_in[39];      // (128,256)
  const float* shot_b = (const float*)d_in[40];

  // ---- workspace bump allocator ----
  char* wsb = (char*)d_ws;
  size_t off = 0;
  auto alloc = [&](size_t bytes) -> void* {
    void* p = wsb + off;
    off += (bytes + 255) & ~(size_t)255;
    return p;
  };
  u16* wb_in   = (u16*)alloc((size_t)512 * 128 * 2);
  u16* wb_out  = (u16*)alloc((size_t)128 * 256 * 2);
  u16* wb_dn   = (u16*)alloc((size_t)64 * 128 * 2);
  u16* wb_up   = (u16*)alloc((size_t)64 * 128 * 2);
  u16* wb_c1   = (u16*)alloc((size_t)128 * 128 * 2);
  u16* wb_c3   = (u16*)alloc((size_t)128 * 128 * 2);
  u16* wb_sin  = (u16*)alloc((size_t)256 * 64 * 2);
  u16* wb_sout = (u16*)alloc((size_t)64 * 128 * 2);
  u16* wb_shot = (u16*)alloc((size_t)128 * 256 * 2);
  u16*   data_bf = (u16*)  alloc((size_t)Bb * C * N * 2);
  float* dataT   = (float*)alloc((size_t)M * C * 4);       // (B,N,C) f32
  u16*   cat_bf  = (u16*)  alloc((size_t)M * 2 * C * 2);   // (B,N,2C) bf16
  u16*   xln_bf  = (u16*)  alloc((size_t)M * C * 2);
  float* xz      = (float*)alloc((size_t)M * 2 * DI * 4);  // 64MB; later aliased x_up/fuse
  float* xc      = (float*)alloc((size_t)M * DI * 4);
  u16*   xc_bf   = (u16*)  alloc((size_t)M * DI * 2);
  float* xdbl    = (float*)alloc((size_t)M * (DR + 16) * 4);
  float* dtb     = (float*)alloc((size_t)M * DI * 4);      // later aliased y_bf
  float* yb      = (float*)alloc((size_t)M * DI * 4);      // 32MB; later aliased phase-3 bufs
  float* mo      = (float*)alloc((size_t)M * C * 4);
  float* x12     = (float*)alloc((size_t)M * C * 4);
  float* st_m  = (float*)alloc((size_t)Bb * C * 4);
  float* st_v  = (float*)alloc((size_t)Bb * C * 4);
  float* st_bv = (float*)alloc((size_t)C * 4);
  float* st2_m = (float*)alloc((size_t)Bb * C * 4);
  float* st2_v = (float*)alloc((size_t)Bb * C * 4);
  float* st2_bv= (float*)alloc((size_t)C * 4);
  float* x_down = (float*)alloc((size_t)Bb * C * K * 4);
  u16*   c1n_pm = (u16*)  alloc((size_t)Bb * K * C * 2);
  float* o1_pm  = (float*)alloc((size_t)Bb * K * C * 4);
  float* out1   = (float*)alloc((size_t)Bb * C * K * 4);
  u16*   outn_bf= (u16*)  alloc((size_t)M2 * DM2 * 2);
  float* sxz    = (float*)alloc((size_t)M2 * 2 * DI2 * 4);
  float* sxc    = (float*)alloc((size_t)M2 * DI2 * 4);
  u16*   sxc_bf = (u16*)  alloc((size_t)M2 * DI2 * 2);
  float* sxdbl  = (float*)alloc((size_t)M2 * (DR2 + 16) * 4);
  float* sdt    = (float*)alloc((size_t)M2 * DI2 * 4);
  float* sy     = (float*)alloc((size_t)M2 * DI2 * 4);
  u16*   sy_bf  = (u16*)  alloc((size_t)M2 * DI2 * 2);
  float* samo   = (float*)alloc((size_t)M2 * DM2 * 4);
  float* out2   = (float*)alloc((size_t)M2 * DM2 * 4);
  u16*   c3n_pm = (u16*)  alloc((size_t)Bb * K * C * 2);
  float* x2pm   = (float*)alloc((size_t)Bb * K * C * 4);
  float* x2t    = (float*)alloc((size_t)Bb * C * K * 4);
  u16*   x2_bf  = (u16*)  alloc((size_t)Bb * C * K * 2);
  // aliases (lifetimes disjoint):
  u16*   y_bf    = (u16*)dtb;                              // dt dead after scan
  char*  ybb     = (char*)yb;                              // y dead after out_proj
  u16*   norm_pm = (u16*)(ybb);                            // 8MB
  float* convo   = (float*)(ybb + (size_t)8  * 1024 * 1024); // 8MB
  u16*   SdnT    = (u16*)  (ybb + (size_t)16 * 1024 * 1024); // 4MB
  u16*   Sup_bf  = (u16*)  (ybb + (size_t)20 * 1024 * 1024); // 4MB
  float* x_up = (float*)xz;                                // xz dead after scan
  float* fuse = (float*)((char*)xz + (size_t)16 * 1024 * 1024);

  auto cdiv = [](long a, long b) -> unsigned { return (unsigned)((a + b - 1) / b); };
  auto gemm = [&](const u16* A, int lda, long sA, const u16* W, int ldb, long sB,
                  float* O, int ldc, long sC, long Mm, int Nn, int Kk,
                  const float* bias, int batch) {
    dim3 g((unsigned)((Mm / 16) * (Nn / 16)), (unsigned)batch);
    k_wmma_gemm<<<g, 32, 0, stream>>>(A, lda, sA, W, ldb, sB, O, ldc, sC, Nn / 16, Kk, bias);
  };

  // ---- phase 0: bf16 conversions ----
  k_f32_to_bf16<<<cdiv(512 * 128, 256), 256, 0, stream>>>(mm_in_w, wb_in, 512 * 128);
  k_f32_to_bf16<<<cdiv(128 * 256, 256), 256, 0, stream>>>(mm_out_w, wb_out, 128 * 256);
  k_f32_to_bf16<<<cdiv(64 * 128, 256), 256, 0, stream>>>(dn_w, wb_dn, 64 * 128);
  k_f32_to_bf16<<<cdiv(64 * 128, 256), 256, 0, stream>>>(up_w, wb_up, 64 * 128);
  k_f32_to_bf16<<<cdiv(128 * 128, 256), 256, 0, stream>>>(c1_w, wb_c1, 128 * 128);
  k_f32_to_bf16<<<cdiv(128 * 128, 256), 256, 0, stream>>>(c3_w, wb_c3, 128 * 128);
  k_f32_to_bf16<<<cdiv(256 * 64, 256), 256, 0, stream>>>(sm_in_w, wb_sin, 256 * 64);
  k_f32_to_bf16<<<cdiv(64 * 128, 256), 256, 0, stream>>>(sm_out_w, wb_sout, 64 * 128);
  k_f32_to_bf16<<<cdiv(128 * 256, 256), 256, 0, stream>>>(shot_w, wb_shot, 128 * 256);
  k_f32_to_bf16<<<cdiv((long)Bb * C * N, 256), 256, 0, stream>>>(data, data_bf, (long)Bb * C * N);

  // ---- phase 1: data (B,C,N) -> dataT (B,N,C) f32 + cat[:, :C] bf16 ----
  { dim3 g(cdiv(C, 16), cdiv(N, 16), Bb), blk(16, 16);
    k_transpose<<<g, blk, 0, stream>>>(data, C, N, dataT, C, cat_bf, 2 * C, 0); }

  // ---- phase 2: Mamba block over N (d_model=C) ----
  k_layernorm<<<(unsigned)M, 128, 0, stream>>>(dataT, nullptr, mb_ln_w, mb_ln_b, nullptr, xln_bf, C);
  gemm(xln_bf, C, 0, wb_in, C, 0, xz, 2 * DI, 0, M, 2 * DI, C, nullptr, 1);
  k_dwconv_silu<<<cdiv(Bb * DI, 256), 256, 0, stream>>>(xz, 2 * DI, mm_conv_w, mm_conv_b,
                                                        xc, xc_bf, Bb, N, DI);
  k_gemm_small<<<cdiv(M * 24, 256), 256, 0, stream>>>(xc, DI, mm_xproj_w, DI, nullptr,
                                                      xdbl, 24, M, 24, DI, 0);
  k_gemm_small<<<cdiv(M * DI, 256), 256, 0, stream>>>(xdbl, 24, mm_dtproj_w, DR, mm_dtproj_b,
                                                      dtb, DI, M, DI, DR, 1);
  k_scan<<<cdiv(Bb * DI, 256), 256, 0, stream>>>(dtb, xdbl, 24, DR, xc, xz + DI, 2 * DI,
                                                 mm_Alog, mm_D, yb, Bb, N, DI);
  k_f32_to_bf16<<<cdiv(M * DI, 256), 256, 0, stream>>>(yb, y_bf, M * DI);
  gemm(y_bf, DI, 0, wb_out, DI, 0, mo, C, 0, M, C, DI, nullptr, 1);
  k_layernorm<<<(unsigned)M, 128, 0, stream>>>(mo, dataT, mb_ln_w, mb_ln_b, x12, nullptr, C);

  // ---- phase 3: diff_pool (down) ----
  k_inorm_stats<<<Bb * C, 256, 0, stream>>>(data, N, st_m, st_v);
  k_bnorm_var<<<1, 128, 0, stream>>>(st_v, Bb, C, st_bv);
  k_in_bn_relu<<<(unsigned)M, C, 0, stream>>>(data, N, C, st_m, st_v, st_bv, dn_bnw, dn_bnb, norm_pm);
  gemm(norm_pm, C, 0, wb_dn, C, 0, convo, K, 0, M, K, C, dn_b, 1);
  k_softmax<<<Bb * K, 256, 0, stream>>>(convo, (long)N * K, K, 1L, N, K);   // over n
  { dim3 g(cdiv(N, 16), cdiv(K, 16), Bb), blk(16, 16);
    k_transpose<<<g, blk, 0, stream>>>(convo, N, K, nullptr, 0, SdnT, N, 0); }
  gemm(data_bf, N, (long)C * N, SdnT, N, (long)K * N, x_down, K, (long)C * K, C, K, N, nullptr, Bb);

  // ---- phase 4: CSM (conv1 -> LN -> SAM -> conv3) ----
  k_inorm_stats<<<Bb * C, 256, 0, stream>>>(x_down, K, st2_m, st2_v);
  k_bnorm_var<<<1, 128, 0, stream>>>(st2_v, Bb, C, st2_bv);
  k_in_bn_relu<<<Bb * K, C, 0, stream>>>(x_down, K, C, st2_m, st2_v, st2_bv, c1_bnw, c1_bnb, c1n_pm);
  gemm(c1n_pm, C, 0, wb_c1, C, 0, o1_pm, C, 0, (long)Bb * K, C, C, c1_b, 1);
  { dim3 g(cdiv(K, 16), cdiv(C, 16), Bb), blk(16, 16);
    k_transpose<<<g, blk, 0, stream>>>(o1_pm, K, C, out1, K, nullptr, 0, 0); }
  k_layernorm<<<(unsigned)M2, 128, 0, stream>>>(out1, nullptr, cs_lnw, cs_lnb, nullptr, outn_bf, DM2);
  // SAM mamba over C (d_model=K)
  gemm(outn_bf, DM2, 0, wb_sin, DM2, 0, sxz, 2 * DI2, 0, M2, 2 * DI2, DM2, nullptr, 1);
  k_dwconv_silu<<<cdiv(Bb * DI2, 256), 256, 0, stream>>>(sxz, 2 * DI2, sm_conv_w, sm_conv_b,
                                                         sxc, sxc_bf, Bb, L2, DI2);
  k_gemm_small<<<cdiv(M2 * 20, 256), 256, 0, stream>>>(sxc, DI2, sm_xproj_w, DI2, nullptr,
                                                       sxdbl, 20, M2, 20, DI2, 0);
  k_gemm_small<<<cdiv(M2 * DI2, 256), 256, 0, stream>>>(sxdbl, 20, sm_dtproj_w, DR2, sm_dtproj_b,
                                                        sdt, DI2, M2, DI2, DR2, 1);
  k_scan<<<cdiv(Bb * DI2, 256), 256, 0, stream>>>(sdt, sxdbl, 20, DR2, sxc, sxz + DI2, 2 * DI2,
                                                  sm_Alog, sm_D, sy, Bb, L2, DI2);
  k_f32_to_bf16<<<cdiv(M2 * DI2, 256), 256, 0, stream>>>(sy, sy_bf, M2 * DI2);
  gemm(sy_bf, DI2, 0, wb_sout, DI2, 0, samo, DM2, 0, M2, DM2, DI2, nullptr, 1);
  k_add<<<cdiv(M2 * DM2, 256), 256, 0, stream>>>(out1, samo, out2, M2 * DM2);
  // conv3 + residual x_down
  k_inorm_stats<<<Bb * C, 256, 0, stream>>>(out2, K, st2_m, st2_v);
  k_bnorm_var<<<1, 128, 0, stream>>>(st2_v, Bb, C, st2_bv);
  k_in_bn_relu<<<Bb * K, C, 0, stream>>>(out2, K, C, st2_m, st2_v, st2_bv, c3_bnw, c3_bnb, c3n_pm);
  gemm(c3n_pm, C, 0, wb_c3, C, 0, x2pm, C, 0, (long)Bb * K, C, C, c3_b, 1);
  { dim3 g(cdiv(K, 16), cdiv(C, 16), Bb), blk(16, 16);
    k_transpose<<<g, blk, 0, stream>>>(x2pm, K, C, x2t, K, nullptr, 0, 0); }
  k_add<<<cdiv((long)Bb * C * K, 256), 256, 0, stream>>>(x2t, x_down, x2t, (long)Bb * C * K);
  k_f32_to_bf16<<<cdiv((long)Bb * C * K, 256), 256, 0, stream>>>(x2t, x2_bf, (long)Bb * C * K);

  // ---- phase 5: diff_unpool (up) ----
  k_in_bn_relu<<<(unsigned)M, C, 0, stream>>>(data, N, C, st_m, st_v, st_bv, up_bnw, up_bnb, norm_pm);
  gemm(norm_pm, C, 0, wb_up, C, 0, convo, K, 0, M, K, C, up_b, 1);
  k_softmax<<<(unsigned)M, 256, 0, stream>>>(convo, 0L, 1 << 30, (long)K, K, 1); // over k
  k_f32_to_bf16<<<cdiv(M * K, 256), 256, 0, stream>>>(convo, Sup_bf, M * K);
  gemm(x2_bf, K, (long)C * K, Sup_bf, K, (long)N * K, x_up, N, (long)C * N, C, N, K, nullptr, Bb);

  // ---- phase 6: fuse ----
  { dim3 g(cdiv(C, 16), cdiv(N, 16), Bb), blk(16, 16);
    k_transpose<<<g, blk, 0, stream>>>(x_up, C, N, nullptr, 0, cat_bf, 2 * C, C); }
  gemm(cat_bf, 2 * C, 0, wb_shot, 2 * C, 0, fuse, C, 0, M, C, 2 * C, shot_b, 1);
  k_final<<<cdiv(M * C, 256), 256, 0, stream>>>(fuse, x12, (float*)d_out, Bb, C, N);
}
